// GraphMolActorCritic_thv1_65438121722208
// MI455X (gfx1250) — matrix operand order repaired and checked by reading, compiled, not run on hardware
//
#include <hip/hip_runtime.h>

// ---------------------------------------------------------------------------
// MI455X (gfx1250) GraphMolActorCritic forward.
// - ew [16384,64,64] materialized in bf16 (128 MB < 192 MB L2): written once by
//   a WMMA GEMM, streamed from L2 by all 6 message-passing iterations.
// - All dense layers: v_wmma_f32_16x16x32_bf16, bf16 A buffers + pre-packed
//   bf16 B fragments -> only b128 loads in the GEMM inner loop, fp32 accum.
// - GEMM epilogue fully specialized via template flags (no runtime null checks,
//   incremental 32-bit row addressing).
// - Scatter-mean via global_atomic_add_f32; heads + degenerate Set2Set in
//   small VALU kernels.
// ---------------------------------------------------------------------------

typedef __attribute__((ext_vector_type(16))) __bf16 v16bf;
typedef __attribute__((ext_vector_type(8)))  float  v8f;

#define DIM       64
#define NNODES    8192
#define NEDGES    16384
#define NGRAPHS   256
#define NSTEMS    2048
#define NJBONDS   1024
#define NUMOUT    105
#define NUMFEAT   14

static __device__ __forceinline__ float lrelu_f(float v) { return v > 0.f ? v : 0.01f * v; }
static __device__ __forceinline__ float sigm_f(float v)  { return 1.f / (1.f + __expf(-v)); }
static __device__ __forceinline__ unsigned short f2bf(float f) {
  __bf16 b = (__bf16)f; return __builtin_bit_cast(unsigned short, b);
}
// bf16 decode from packed dwords: lo = shift, hi = mask (1 VALU op each)
static __device__ __forceinline__ float bflo(unsigned u) {
  union { unsigned a; float f; } v; v.a = u << 16; return v.f;
}
static __device__ __forceinline__ float bfhi(unsigned u) {
  union { unsigned a; float f; } v; v.a = u & 0xffff0000u; return v.f;
}

// ---------------------------------------------------------------------------
// Pack f32 weight B[Ksrc,N] into per-lane WMMA B fragments, zero-padded to
// KP (multiple of 32). Layout: Bp[(((nt*(KP/32)+c)*32)+lane)*16 + j]
//   lane<16 : col = nt*16+lane,    k = c*32 + (j&7) + (j>=8 ? 16 : 0)      (j=0..15)
//   lane>=16: col = nt*16+lane-16, k = c*32 + 8 + (j&7) + (j>=8 ? 16 : 0)
// ---------------------------------------------------------------------------
__global__ void k_packB(const float* __restrict__ B, unsigned short* __restrict__ Bp,
                        int N, int Ksrc, int KP)
{
  int t = blockIdx.x * blockDim.x + threadIdx.x;
  int ntiles = (N + 15) / 16;
  int total = ntiles * (KP / 32) * 32 * 16;
  if (t >= total) return;
  int j = t & 15;
  int r = t >> 4;
  int lane = r & 31; r >>= 5;
  int c  = r % (KP / 32);
  int nt = r / (KP / 32);
  int koffB = (lane & 16) ? 8 : 0;
  int col = nt * 16 + (lane & 15);
  int kk  = c * 32 + koffB + (j & 7) + ((j & 8) ? 16 : 0);
  float v = (col < N && kk < Ksrc) ? B[(size_t)kk * N + col] : 0.f;
  Bp[t] = f2bf(v);
}

// ---------------------------------------------------------------------------
// bf16 WMMA GEMM: C = act(A[M,K]bf16 @ B + bias [+ D]). One wave per 16x16 C
// tile, K compile-time (32 or 64) -> fully unrolled, all loads are b128.
// Epilogue fully specialized via template flags.
// A fragment (ISA 7.12.2): lane<16 row=lane koff=0; lane>=16 row=lane-16 koff=8
//   elements [0..7] = K {c*32+koff..+7}, [8..15] = K {c*32+16+koff..+7}
// ---------------------------------------------------------------------------
template<int K, bool HAS_D, bool HAS_CF, bool HAS_CB, bool ACT>
__global__ void k_gemm(const unsigned short* __restrict__ A,
                       const unsigned short* __restrict__ Bp,
                       const float* __restrict__ bias, const float* __restrict__ D,
                       float* __restrict__ Cf, unsigned short* __restrict__ Cb,
                       int M, int N)
{
  const int lane = threadIdx.x & 31;
  const int wave = threadIdx.x >> 5;
  const int mt   = blockIdx.x;
  const int nt   = blockIdx.y * 4 + wave;            // 4 waves / block
  if (nt * 16 >= N) return;                          // whole-wave uniform exit

  const int lrow = lane & 15;
  const int hi   = lane >> 4;
  const unsigned short* Ar = A + (size_t)(mt * 16 + lrow) * K + (hi ? 8 : 0);
  const unsigned short* Bb = Bp + ((size_t)(nt * (K / 32)) * 32 + lane) * 16;

  union V { v16bf v; uint4 u[2]; };
  v8f acc = {};
#pragma unroll
  for (int c = 0; c < K / 32; ++c) {
    V a, b;
    a.u[0] = *(const uint4*)(Ar + c * 32);           // K {c*32+koff .. +7}
    a.u[1] = *(const uint4*)(Ar + c * 32 + 16);      // K {c*32+16+koff .. +7}
    b.u[0] = *(const uint4*)(Bb + (size_t)c * 32 * 16);
    b.u[1] = *(const uint4*)(Bb + (size_t)c * 32 * 16 + 8);
    acc = __builtin_amdgcn_wmma_f32_16x16x32_bf16(false, a.v, false, b.v,
                                                  (short)0, acc, false, false);
  }

  const int bcol = nt * 16 + lrow;
  if (bcol >= N) return;
  const float    bv   = bias[bcol];
  unsigned       base = (unsigned)(mt * 16 + hi * 8) * (unsigned)N + (unsigned)bcol;
  const unsigned strd = (unsigned)N;
#pragma unroll
  for (int r = 0; r < 8; ++r) {
    float v = acc[r] + bv;
    if constexpr (HAS_D)  v += D[base];
    if constexpr (ACT)    v = lrelu_f(v);
    if constexpr (HAS_CF) Cf[base] = v;
    if constexpr (HAS_CB) Cb[base] = f2bf(v);
    base += strd;
  }
}

// ---------------------------------------------------------------------------
// Small kernels
// ---------------------------------------------------------------------------
__global__ void k_zero(float* __restrict__ p, int n) {
  int t = blockIdx.x * blockDim.x + threadIdx.x;
  if (t < n) p[t] = 0.f;
}

__global__ void k_deg(const int* __restrict__ dst, float* __restrict__ deg) {
  int e = blockIdx.x * blockDim.x + threadIdx.x;
  if (e < NEDGES) atomicAdd(deg + dst[e], 1.0f);
}

__global__ void k_count(const int* __restrict__ batch, int* __restrict__ counts) {
  int n = blockIdx.x * blockDim.x + threadIdx.x;
  if (n < NNODES) atomicAdd(counts + batch[n], 1);
}

__global__ void k_scan(const int* __restrict__ counts, int* __restrict__ offs) {
  if (threadIdx.x == 0 && blockIdx.x == 0) {
    int a = 0;
    for (int g = 0; g < NGRAPHS; ++g) { offs[g] = a; a += counts[g]; }
  }
}

// x [N,14] f32 -> x_bf [N,32] bf16 zero-padded (K=32 for lin0 WMMA GEMM)
__global__ void k_padx(const float* __restrict__ x, unsigned short* __restrict__ xb) {
  int t = blockIdx.x * blockDim.x + threadIdx.x;
  if (t >= NNODES * 32) return;
  int n = t >> 5, c = t & 31;
  xb[t] = f2bf(c < NUMFEAT ? x[(size_t)n * NUMFEAT + c] : 0.f);
}

// h2 = lrelu(edge_attr[E,4] @ net_w1[4,64] + net_b1) -> bf16 (A of ew GEMM)
__global__ void k_edgenet1(const float* __restrict__ ea, const float* __restrict__ w1,
                           const float* __restrict__ b1, unsigned short* __restrict__ h2)
{
  int t = blockIdx.x * blockDim.x + threadIdx.x;
  if (t >= NEDGES * DIM) return;
  int e = t >> 6, o = t & 63;
  const float* a = ea + (size_t)e * 4;
  float v = b1[o] + a[0] * w1[o] + a[1] * w1[64 + o] + a[2] * w1[128 + o] + a[3] * w1[192 + o];
  h2[t] = f2bf(lrelu_f(v));
}

// Per-edge GEMV msg = out[src] @ ew_e, scatter-add into agg.
// One wave / edge: lanes = 4 rows x 8 column-groups, b128 ew loads (L2-hot),
// fp32 accumulate, 2x shfl_xor cross-lane reduce, 64 float atomics per edge.
__global__ void k_msg(const unsigned short* __restrict__ out_bf,
                      const unsigned short* __restrict__ ew,
                      const int* __restrict__ src, const int* __restrict__ dst,
                      float* __restrict__ agg)
{
  __shared__ float arow[8][DIM];
  const int lane = threadIdx.x & 31;
  const int wave = threadIdx.x >> 5;
  const int e    = blockIdx.x * 8 + wave;

  const int s = src[e];
  unsigned u = ((const unsigned*)(out_bf + (size_t)s * DIM))[lane];
  arow[wave][lane * 2 + 0] = bflo(u);
  arow[wave][lane * 2 + 1] = bfhi(u);
  __syncthreads();

  // CDNA5 prefetch (global_prefetch_b8) of the next block's ew rows
  __builtin_prefetch((const void*)(ew + (size_t)(e + 8) * (DIM * DIM)), 0, 0);

  const int rowOff = lane >> 3;        // 0..3
  const int colg   = lane & 7;         // columns colg*8 .. colg*8+7
  const uint4* pe  = (const uint4*)(ew + (size_t)e * (DIM * DIM));  // 8 uint4 / row
  float acc[8] = {0.f, 0.f, 0.f, 0.f, 0.f, 0.f, 0.f, 0.f};
#pragma unroll
  for (int i0 = 0; i0 < DIM; i0 += 4) {
    const int   row = i0 + rowOff;
    const float ai  = arow[wave][row];
    uint4 w = pe[row * 8 + colg];
    acc[0] = __builtin_fmaf(ai, bflo(w.x), acc[0]);
    acc[1] = __builtin_fmaf(ai, bfhi(w.x), acc[1]);
    acc[2] = __builtin_fmaf(ai, bflo(w.y), acc[2]);
    acc[3] = __builtin_fmaf(ai, bfhi(w.y), acc[3]);
    acc[4] = __builtin_fmaf(ai, bflo(w.z), acc[4]);
    acc[5] = __builtin_fmaf(ai, bfhi(w.z), acc[5]);
    acc[6] = __builtin_fmaf(ai, bflo(w.w), acc[6]);
    acc[7] = __builtin_fmaf(ai, bfhi(w.w), acc[7]);
  }
#pragma unroll
  for (int k = 0; k < 8; ++k) {        // reduce lanes {colg, colg+8, +16, +24}
    acc[k] += __shfl_xor(acc[k], 8, 32);
    acc[k] += __shfl_xor(acc[k], 16, 32);
  }
  if (lane < 8) {
    float* ap = agg + (size_t)dst[e] * DIM + colg * 8;
#pragma unroll
    for (int k = 0; k < 8; ++k) atomicAdd(ap + k, acc[k]);
  }
}

// aggn = agg / max(deg,1)
__global__ void k_divdeg(const float* __restrict__ agg, const float* __restrict__ deg,
                         float* __restrict__ aggn)
{
  int t = blockIdx.x * blockDim.x + threadIdx.x;
  if (t >= NNODES * DIM) return;
  int n = t >> 6;
  aggn[t] = agg[t] / fmaxf(deg[n], 1.f);
}

// GRU update (torch gate order r,z,n); writes h (f32) and its bf16 mirror.
__global__ void k_gru(const float* __restrict__ gi, const float* __restrict__ gh,
                      float* __restrict__ h, unsigned short* __restrict__ out_bf)
{
  int t = blockIdx.x * blockDim.x + threadIdx.x;
  if (t >= NNODES * DIM) return;
  int n = t >> 6, c = t & 63;
  const float* pi = gi + (size_t)n * 3 * DIM;
  const float* ph = gh + (size_t)n * 3 * DIM;
  float r  = sigm_f(pi[c] + ph[c]);
  float z  = sigm_f(pi[DIM + c] + ph[DIM + c]);
  float nn = tanhf(pi[2 * DIM + c] + r * ph[2 * DIM + c]);
  float hv = h[t];
  hv = (1.f - z) * nn + z * hv;
  h[t] = hv;
  out_bf[t] = f2bf(hv);
}

// dst[r,:] = bf16(h[idx[r],:])
__global__ void k_gather_bf(const float* __restrict__ h, const int* __restrict__ idx,
                            unsigned short* __restrict__ dstb, int rows)
{
  int t = blockIdx.x * blockDim.x + threadIdx.x;
  if (t >= rows * DIM) return;
  int r = t >> 6, o = t & 63;
  dstb[t] = f2bf(h[(size_t)idx[r] * DIM + o]);
}

// jbond head tail: scalar per endpoint then mean over the pair
__global__ void k_jpred(const float* __restrict__ j1, const float* __restrict__ w2,
                        const float* __restrict__ b2, float* __restrict__ outj)
{
  int j = blockIdx.x * blockDim.x + threadIdx.x;
  if (j >= NJBONDS) return;
  const float* r0 = j1 + (size_t)(2 * j) * DIM;
  const float* r1 = r0 + DIM;
  float p0 = b2[0], p1 = b2[0];
#pragma unroll 8
  for (int o = 0; o < DIM; ++o) { p0 += r0[o] * w2[o]; p1 += r1[o] * w2[o]; }
  outj[j] = 0.5f * (p0 + p1);
}

// Set2Set step-1 query: zero state -> gates are pure bias -> one shared q.
__global__ void k_q(const float* __restrict__ b_ih, const float* __restrict__ b_hh,
                    float* __restrict__ q)
{
  int o = threadIdx.x;
  if (o < DIM) {
    float gi_ = b_ih[o] + b_hh[o];
    float gg_ = b_ih[2 * DIM + o] + b_hh[2 * DIM + o];
    float go_ = b_ih[3 * DIM + o] + b_hh[3 * DIM + o];
    float c   = sigm_f(gi_) * tanhf(gg_);
    q[o]      = sigm_f(go_) * tanhf(c);
  }
}

// e[n] = dot(h[n], q)
__global__ void k_escore(const float* __restrict__ h, const float* __restrict__ q,
                         float* __restrict__ evec)
{
  int n = blockIdx.x * blockDim.x + threadIdx.x;
  if (n >= NNODES) return;
  const float* r = h + (size_t)n * DIM;
  float s = 0.f;
#pragma unroll 8
  for (int o = 0; o < DIM; ++o) s += r[o] * q[o];
  evec[n] = s;
}

// Per-graph softmax pooling + scalar head. One block (64 threads) per graph.
__global__ void k_pool(const float* __restrict__ h, const float* __restrict__ evec,
                       const float* __restrict__ q, const int* __restrict__ offs,
                       const int* __restrict__ cnts, const float* __restrict__ lw,
                       const float* __restrict__ lb, float* __restrict__ outp)
{
  __shared__ float red[DIM];
  const int g = blockIdx.x, t = threadIdx.x;
  const int beg = offs[g], n = cnts[g];

  float mx = -1e30f;
  for (int i = t; i < n; i += DIM) mx = fmaxf(mx, evec[beg + i]);
  red[t] = mx; __syncthreads();
  for (int s = 32; s > 0; s >>= 1) { if (t < s) red[t] = fmaxf(red[t], red[t + s]); __syncthreads(); }
  mx = red[0]; __syncthreads();

  float sm = 0.f;
  for (int i = t; i < n; i += DIM) sm += __expf(evec[beg + i] - mx);
  red[t] = sm; __syncthreads();
  for (int s = 32; s > 0; s >>= 1) { if (t < s) red[t] += red[t + s]; __syncthreads(); }
  sm = red[0]; __syncthreads();
  const float inv = (sm > 0.f) ? 1.f / sm : 0.f;

  float r = 0.f;                                      // r_pool component o = t
  for (int i = 0; i < n; ++i) {
    float w = __expf(evec[beg + i] - mx) * inv;
    r += w * h[(size_t)(beg + i) * DIM + t];
  }

  const float qo = q[t];
  float p0 = qo * lw[t * 2 + 0] + r * lw[(DIM + t) * 2 + 0];
  float p1 = qo * lw[t * 2 + 1] + r * lw[(DIM + t) * 2 + 1];
  red[t] = p0; __syncthreads();
  for (int s = 32; s > 0; s >>= 1) { if (t < s) red[t] += red[t + s]; __syncthreads(); }
  if (t == 0) outp[g * 2 + 0] = red[0] + lb[0];
  __syncthreads(); red[t] = p1; __syncthreads();
  for (int s = 32; s > 0; s >>= 1) { if (t < s) red[t] += red[t + s]; __syncthreads(); }
  if (t == 0) outp[g * 2 + 1] = red[0] + lb[1];
}

// ---------------------------------------------------------------------------
// Host-side orchestration
// ---------------------------------------------------------------------------
template<int K, bool HAS_D, bool HAS_CF, bool HAS_CB, bool ACT>
static void gemmT(hipStream_t s, const unsigned short* A, const unsigned short* Bp,
                  const float* bias, const float* D, float* Cf, unsigned short* Cb,
                  int M, int N)
{
  dim3 grid(M / 16, ((N + 15) / 16 + 3) / 4);
  k_gemm<K, HAS_D, HAS_CF, HAS_CB, ACT><<<grid, 128, 0, s>>>(A, Bp, bias, D, Cf, Cb, M, N);
}

static void packB(hipStream_t s, const float* B, unsigned short* Bp, int N, int Ksrc, int KP)
{
  int total = ((N + 15) / 16) * (KP / 32) * 32 * 16;
  k_packB<<<(total + 255) / 256, 256, 0, s>>>(B, Bp, N, Ksrc, KP);
}

static inline size_t packSize(int N, int KP) {           // in ushorts
  return (size_t)((N + 15) / 16) * (KP / 32) * 32 * 16;
}

extern "C" void kernel_launch(void* const* d_in, const int* in_sizes, int n_in,
                              void* d_out, int out_size, void* d_ws, size_t ws_size,
                              hipStream_t stream)
{
  (void)in_sizes; (void)n_in; (void)out_size; (void)ws_size;

  // Inputs in setup_inputs() dict order
  const float* x          = (const float*)d_in[0];
  const float* edge_attr  = (const float*)d_in[1];
  const int*   edge_index = (const int*)d_in[2];
  const int*   batch      = (const int*)d_in[3];
  const int*   stem_idx   = (const int*)d_in[4];
  const int*   jbond_idx  = (const int*)d_in[5];
  const float* lin0_w     = (const float*)d_in[6];
  const float* lin0_b     = (const float*)d_in[7];
  const float* net_w1     = (const float*)d_in[8];
  const float* net_b1     = (const float*)d_in[9];
  const float* net_w2     = (const float*)d_in[10];
  const float* net_b2     = (const float*)d_in[11];
  const float* conv_root  = (const float*)d_in[12];
  const float* conv_bias  = (const float*)d_in[13];
  const float* gru_w_ih   = (const float*)d_in[14];
  const float* gru_w_hh   = (const float*)d_in[15];
  const float* gru_b_ih   = (const float*)d_in[16];
  const float* gru_b_hh   = (const float*)d_in[17];
  const float* n2s_w1     = (const float*)d_in[18];
  const float* n2s_b1     = (const float*)d_in[19];
  const float* n2s_w2     = (const float*)d_in[20];
  const float* n2s_b2     = (const float*)d_in[21];
  const float* n2j_w1     = (const float*)d_in[22];
  const float* n2j_b1     = (const float*)d_in[23];
  const float* n2j_w2     = (const float*)d_in[24];
  const float* n2j_b2     = (const float*)d_in[25];
  const float* s2s_b_ih   = (const float*)d_in[28];
  const float* s2s_b_hh   = (const float*)d_in[29];
  const float* lin_out_w  = (const float*)d_in[30];
  const float* lin_out_b  = (const float*)d_in[31];

  const int* src = edge_index;            // edge_index[0]
  const int* dst = edge_index + NEDGES;   // edge_index[1]

  float* out_scalar = (float*)d_out;                  // [256,2]
  float* out_stem   = out_scalar + NGRAPHS * 2;       // [2048,105]
  float* out_jbond  = out_stem + NSTEMS * NUMOUT;     // [1024]

  // Workspace bump allocator (~152 MB; ew_bf 128 MB is L2-resident by design)
  char* wp = (char*)d_ws;
  auto alloc = [&](size_t bytes) -> char* {
    char* p = wp; wp += (bytes + 255) & ~(size_t)255; return p;
  };
  unsigned short* ew_bf  = (unsigned short*)alloc((size_t)NEDGES * DIM * DIM * 2);
  unsigned short* h2_bf  = (unsigned short*)alloc((size_t)NEDGES * DIM * 2);
  unsigned short* x_bf   = (unsigned short*)alloc((size_t)NNODES * 32 * 2);
  float*          hbuf   = (float*)alloc((size_t)NNODES * DIM * 4);
  unsigned short* out_bf = (unsigned short*)alloc((size_t)NNODES * DIM * 2);
  unsigned short* m_bf   = (unsigned short*)alloc((size_t)NNODES * DIM * 2);
  float*          gi     = (float*)alloc((size_t)NNODES * 3 * DIM * 4);
  float*          gh     = (float*)alloc((size_t)NNODES * 3 * DIM * 4);
  float*          agg    = (float*)alloc((size_t)NNODES * DIM * 4);
  float*          aggn   = (float*)alloc((size_t)NNODES * DIM * 4);
  float*          deg    = (float*)alloc((size_t)NNODES * 4);
  unsigned short* stem_bf= (unsigned short*)alloc((size_t)NSTEMS * DIM * 2);
  unsigned short* s1_bf  = (unsigned short*)alloc((size_t)NSTEMS * DIM * 2);
  unsigned short* jb_bf  = (unsigned short*)alloc((size_t)2 * NJBONDS * DIM * 2);
  float*          j1     = (float*)alloc((size_t)2 * NJBONDS * DIM * 4);
  float*          qv     = (float*)alloc(DIM * 4);
  float*          evec   = (float*)alloc((size_t)NNODES * 4);
  int*            counts = (int*)alloc(NGRAPHS * 4);
  int*            offs   = (int*)alloc(NGRAPHS * 4);
  // Pre-packed bf16 B fragments
  unsigned short* p_lin0 = (unsigned short*)alloc(packSize(DIM, 32) * 2);
  unsigned short* p_w2   = (unsigned short*)alloc(packSize(DIM * DIM, 64) * 2);
  unsigned short* p_root = (unsigned short*)alloc(packSize(DIM, 64) * 2);
  unsigned short* p_gih  = (unsigned short*)alloc(packSize(3 * DIM, 64) * 2);
  unsigned short* p_ghh  = (unsigned short*)alloc(packSize(3 * DIM, 64) * 2);
  unsigned short* p_s1   = (unsigned short*)alloc(packSize(DIM, 64) * 2);
  unsigned short* p_s2   = (unsigned short*)alloc(packSize(NUMOUT, 64) * 2);
  unsigned short* p_j1   = (unsigned short*)alloc(packSize(DIM, 64) * 2);

  const int TB = 256;
  auto blks = [](int n, int t) { return (n + t - 1) / t; };

  // Degree + batch segmentation (fully rewritten every call -> deterministic)
  k_zero<<<blks(NNODES, TB), TB, 0, stream>>>(deg, NNODES);
  k_zero<<<1, TB, 0, stream>>>((float*)counts, NGRAPHS);
  k_deg<<<blks(NEDGES, TB), TB, 0, stream>>>(dst, deg);
  k_count<<<blks(NNODES, TB), TB, 0, stream>>>(batch, counts);
  k_scan<<<1, 32, 0, stream>>>(counts, offs);

  // Pack all GEMM B operands into WMMA fragment layout (tiny, once per call)
  packB(stream, lin0_w,    p_lin0, DIM,       NUMFEAT, 32);
  packB(stream, net_w2,    p_w2,   DIM * DIM, DIM,     64);
  packB(stream, conv_root, p_root, DIM,       DIM,     64);
  packB(stream, gru_w_ih,  p_gih,  3 * DIM,   DIM,     64);
  packB(stream, gru_w_hh,  p_ghh,  3 * DIM,   DIM,     64);
  packB(stream, n2s_w1,    p_s1,   DIM,       DIM,     64);
  packB(stream, n2s_w2,    p_s2,   NUMOUT,    DIM,     64);
  packB(stream, n2j_w1,    p_j1,   DIM,       DIM,     64);

  // lin0: out = lrelu(x @ lin0_w + b); x zero-padded to K=32 in bf16
  k_padx<<<blks(NNODES * 32, TB), TB, 0, stream>>>(x, x_bf);
  gemmT<32, false, true, true, true>(stream, x_bf, p_lin0, lin0_b, nullptr,
                                     hbuf, out_bf, NNODES, DIM);

  // Edge net: h2 = lrelu(ea @ w1 + b1) bf16; ew = h2 @ w2 + b2 -> bf16, L2-res
  k_edgenet1<<<blks(NEDGES * DIM, TB), TB, 0, stream>>>(edge_attr, net_w1, net_b1, h2_bf);
  gemmT<64, false, false, true, false>(stream, h2_bf, p_w2, net_b2, nullptr,
                                       nullptr, ew_bf, NEDGES, DIM * DIM);

  // 6 message-passing + GRU iterations
  for (int it = 0; it < 6; ++it) {
    k_zero<<<blks(NNODES * DIM, TB), TB, 0, stream>>>(agg, NNODES * DIM);
    k_msg<<<NEDGES / 8, 256, 0, stream>>>(out_bf, ew_bf, src, dst, agg);
    k_divdeg<<<blks(NNODES * DIM, TB), TB, 0, stream>>>(agg, deg, aggn);
    // m = lrelu(aggn + out @ conv_root + conv_bias), emitted as bf16 for gi GEMM
    gemmT<64, true, false, true, true>(stream, out_bf, p_root, conv_bias, aggn,
                                       nullptr, m_bf, NNODES, DIM);
    gemmT<64, false, true, false, false>(stream, m_bf, p_gih, gru_b_ih, nullptr,
                                         gi, nullptr, NNODES, 3 * DIM);
    gemmT<64, false, true, false, false>(stream, out_bf, p_ghh, gru_b_hh, nullptr,
                                         gh, nullptr, NNODES, 3 * DIM);
    k_gru<<<blks(NNODES * DIM, TB), TB, 0, stream>>>(gi, gh, hbuf, out_bf);
  }

  // Stem head -> d_out[512 : 512+2048*105]
  k_gather_bf<<<blks(NSTEMS * DIM, TB), TB, 0, stream>>>(hbuf, stem_idx, stem_bf, NSTEMS);
  gemmT<64, false, false, true, true>(stream, stem_bf, p_s1, n2s_b1, nullptr,
                                      nullptr, s1_bf, NSTEMS, DIM);
  gemmT<64, false, true, false, false>(stream, s1_bf, p_s2, n2s_b2, nullptr,
                                       out_stem, nullptr, NSTEMS, NUMOUT);

  // Jbond head -> d_out tail
  k_gather_bf<<<blks(2 * NJBONDS * DIM, TB), TB, 0, stream>>>(hbuf, jbond_idx, jb_bf, 2 * NJBONDS);
  gemmT<64, false, true, false, true>(stream, jb_bf, p_j1, n2j_b1, nullptr,
                                      j1, nullptr, 2 * NJBONDS, DIM);
  k_jpred<<<blks(NJBONDS, TB), TB, 0, stream>>>(j1, n2j_w2, n2j_b2, out_jbond);

  // Set2Set (steps=1, zero state => bias-only query) + scalar head -> d_out[0:512]
  k_q<<<1, DIM, 0, stream>>>(s2s_b_ih, s2s_b_hh, qv);
  k_escore<<<blks(NNODES, TB), TB, 0, stream>>>(hbuf, qv, evec);
  k_pool<<<NGRAPHS, DIM, 0, stream>>>(hbuf, evec, qv, offs, counts, lin_out_w, lin_out_b, out_scalar);
}
